// GATModel_34634616275593
// MI455X (gfx1250) — matrix operand order repaired and checked by reading, not compiled, not run on hardware
//
#include <hip/hip_runtime.h>
#include <hip/hip_bf16.h>

// ---------------------------------------------------------------------------
// GAT (2-layer, heads=2) for MI455X / gfx1250.
//   - Dense projections via V_WMMA_F32_16X16X32_BF16 (fp32 accumulate).
//   - Edge softmax/aggregation via f32 global atomics (L2-resident dst rows).
// ---------------------------------------------------------------------------

typedef __attribute__((ext_vector_type(16))) __bf16 v16bf;
typedef __attribute__((ext_vector_type(8)))  float  v8f;

union BF16x16 {
    unsigned short u[16];
    v16bf v;
};

__device__ __forceinline__ unsigned short f2bf(float f) {
    unsigned u = __float_as_uint(f);
    unsigned r = u + 0x7FFFu + ((u >> 16) & 1u);   // round-to-nearest-even
    return (unsigned short)(r >> 16);
}

// order-preserving float <-> uint for atomicMax over signed floats
__device__ __forceinline__ unsigned ordf(float f) {
    unsigned u = __float_as_uint(f);
    return (u & 0x80000000u) ? ~u : (u | 0x80000000u);
}
__device__ __forceinline__ float unordf(unsigned u) {
    return (u & 0x80000000u) ? __uint_as_float(u ^ 0x80000000u)
                             : __uint_as_float(~u);
}

// ---------------------------------------------------------------------------
// C[nrows x NC] = A[nrows x 128] * W[128 x NC]   (bf16 WMMA, fp32 accum)
// One wave computes 16 rows x NC (NC/16 tiles of 16x16); K=128 -> 4 WMMA steps per tile.
// A fragment kept in registers (contiguous K order, k = 16*(l>>4) + i, same as B); W transposed into
// LDS as bf16 so each lane's 16 K-values are one contiguous 32B chunk.
// ---------------------------------------------------------------------------
__global__ void gemm_bf16_wmma(const float* __restrict__ A,
                               const float* __restrict__ W,
                               float* __restrict__ C,
                               int nrows, int NC) {
    extern __shared__ unsigned short ldsB[];   // [NC][128] : ldsB[c*128 + k]

    const int tid = threadIdx.x;
    // cooperative load + convert + transpose of W (row-major [128][NC])
    for (int g = tid; g < 128 * NC; g += blockDim.x) {
        int k = g / NC, c = g - k * NC;
        ldsB[c * 128 + k] = f2bf(W[g]);
    }
    __syncthreads();

    const int wave = tid >> 5;
    const int lane = tid & 31;
    const int rowTile = blockIdx.x * (blockDim.x >> 5) + wave;
    if (rowTile * 16 >= nrows) return;
    const int rowBase = rowTile * 16;

    // A fragment: lane l -> row = rowBase + (l&15), kbase = 16*(l>>4).
    // element i of v16bf: K = kk*32 + kbase + i
    const int row   = rowBase + (lane & 15);
    const int kbase = (lane >> 4) * 16;
    const float* ar = A + (size_t)row * 128;

    BF16x16 afr[4];
#pragma unroll
    for (int kk = 0; kk < 4; ++kk) {
        float4 lo0 = *(const float4*)(ar + kk * 32 + kbase);
        float4 lo1 = *(const float4*)(ar + kk * 32 + kbase + 4);
        float4 hi0 = *(const float4*)(ar + kk * 32 + kbase + 8);
        float4 hi1 = *(const float4*)(ar + kk * 32 + kbase + 12);
        float t[16] = {lo0.x, lo0.y, lo0.z, lo0.w, lo1.x, lo1.y, lo1.z, lo1.w,
                       hi0.x, hi0.y, hi0.z, hi0.w, hi1.x, hi1.y, hi1.z, hi1.w};
#pragma unroll
        for (int i = 0; i < 16; ++i) afr[kk].u[i] = f2bf(t[i]);
    }

    const int ntiles = NC >> 4;
    for (int ct = 0; ct < ntiles; ++ct) {
        v8f acc = {};
        // B: lane l -> col = ct*16 + (l&15); lanes 0-15 read K[kk*32 .. +15],
        // lanes 16-31 read K[kk*32+16 .. +31] (contiguous 32B in LDS).
        const unsigned short* bp =
            &ldsB[((size_t)(ct * 16 + (lane & 15))) * 128 + ((lane >> 4) * 16)];
#pragma unroll
        for (int kk = 0; kk < 4; ++kk) {
            v16bf b = *(const v16bf*)(bp + kk * 32);
            acc = __builtin_amdgcn_wmma_f32_16x16x32_bf16(
                false, afr[kk].v, false, b, (short)0, acc, false, false);
        }
        // D layout: lane l, vgpr r -> M = r + 8*(l>=16), N = l&15
        float* cp = C + (size_t)(rowBase + (lane >> 4) * 8) * NC
                      + ct * 16 + (lane & 15);
#pragma unroll
        for (int r = 0; r < 8; ++r) cp[(size_t)r * NC] = acc[r];
    }
}

// ---------------------------------------------------------------------------
// a_src[n,h] = sum_c xh[n,h,c]*att_src[h,c]  (same for a_dst). One wave/node:
// lanes 0-15 = head0, 16-31 = head1; shfl_xor reduce within 16-lane halves.
// ---------------------------------------------------------------------------
__global__ void attn_kernel(const float* __restrict__ xh,
                            const float* __restrict__ att_src,
                            const float* __restrict__ att_dst,
                            float* __restrict__ a_src,
                            float* __restrict__ a_dst,
                            int n, int C) {
    const int wave = threadIdx.x >> 5;
    const int lane = threadIdx.x & 31;
    const int node = blockIdx.x * (blockDim.x >> 5) + wave;
    if (node >= n) return;
    const int head = lane >> 4;
    const int ls   = lane & 15;
    const int per  = C >> 4;
    const float* xr = xh + (size_t)node * 2 * C + head * C + ls * per;
    const float* as = att_src + head * C + ls * per;
    const float* ad = att_dst + head * C + ls * per;
    float ss = 0.f, ds = 0.f;
    for (int j = 0; j < per; ++j) {
        float v = xr[j];
        ss += v * as[j];
        ds += v * ad[j];
    }
#pragma unroll
    for (int m = 1; m < 16; m <<= 1) {
        ss += __shfl_xor(ss, m, 32);
        ds += __shfl_xor(ds, m, 32);
    }
    if (ls == 0) {
        a_src[(size_t)node * 2 + head] = ss;
        a_dst[(size_t)node * 2 + head] = ds;
    }
}

__device__ __forceinline__ void edge_endpoints(const int* __restrict__ ei,
                                               int E, int e, int& s, int& d) {
    if (e < E) { s = ei[e]; d = ei[E + e]; }
    else       { s = e - E; d = e - E; }        // self-loops appended
}

// pass 1: segment max of leaky_relu(a_src[s]+a_dst[d]) over dst
__global__ void edge_max_kernel(const int* __restrict__ ei, int E, int n,
                                const float* __restrict__ a_src,
                                const float* __restrict__ a_dst,
                                unsigned* __restrict__ mord) {
    long long idx = (long long)blockIdx.x * blockDim.x + threadIdx.x;
    long long tot = 2LL * (E + n);
    if (idx >= tot) return;
    int e = (int)(idx >> 1), h = (int)(idx & 1);
    int s, d; edge_endpoints(ei, E, e, s, d);
    float t = a_src[(size_t)s * 2 + h] + a_dst[(size_t)d * 2 + h];
    t = (t > 0.f) ? t : 0.2f * t;
    atomicMax(&mord[(size_t)d * 2 + h], ordf(t));
}

// pass 2: exp(e - max) per edge, segment sum into denom
__global__ void edge_exp_kernel(const int* __restrict__ ei, int E, int n,
                                const float* __restrict__ a_src,
                                const float* __restrict__ a_dst,
                                const unsigned* __restrict__ mord,
                                float* __restrict__ expe,
                                float* __restrict__ denom) {
    long long idx = (long long)blockIdx.x * blockDim.x + threadIdx.x;
    long long tot = 2LL * (E + n);
    if (idx >= tot) return;
    int e = (int)(idx >> 1), h = (int)(idx & 1);
    int s, d; edge_endpoints(ei, E, e, s, d);
    float t = a_src[(size_t)s * 2 + h] + a_dst[(size_t)d * 2 + h];
    t = (t > 0.f) ? t : 0.2f * t;
    float m  = unordf(mord[(size_t)d * 2 + h]);
    float ex = __expf(t - m);
    expe[(size_t)e * 2 + h] = ex;
    atomicAdd(&denom[(size_t)d * 2 + h], ex);
}

// pass 3: agg[d] += alpha * xh[s]  (one 64-thread block per edge)
__global__ void edge_agg_kernel(const int* __restrict__ ei, int E, int n,
                                const float* __restrict__ xh,
                                const float* __restrict__ expe,
                                const float* __restrict__ denom,
                                float* __restrict__ agg, int C) {
    int e = blockIdx.x;
    int s, d; edge_endpoints(ei, E, e, s, d);
    const int W = 2 * C;
    for (int c = threadIdx.x; c < W; c += blockDim.x) {
        int h = (c >= C);
        float alpha = expe[(size_t)e * 2 + h] /
                      (denom[(size_t)d * 2 + h] + 1e-16f);
        atomicAdd(&agg[(size_t)d * W + c], xh[(size_t)s * W + c] * alpha);
    }
}

__global__ void bias_relu_kernel(float* __restrict__ buf,
                                 const float* __restrict__ b,
                                 long long total, int width) {
    long long i = (long long)blockIdx.x * blockDim.x + threadIdx.x;
    if (i >= total) return;
    float v = buf[i] + b[(int)(i % width)];
    buf[i] = (v > 0.f) ? v : 0.f;
}

// head-mean + bias + graph mean-pool accumulation (one 128-thread block/node)
__global__ void mean_pool_kernel(const float* __restrict__ agg2,
                                 const float* __restrict__ b2,
                                 const int* __restrict__ batch,
                                 float* __restrict__ pooled,
                                 float* __restrict__ cnt, int n) {
    int node = blockIdx.x;
    if (node >= n) return;
    int c = threadIdx.x;   // 0..127
    float o = 0.5f * (agg2[(size_t)node * 256 + c] +
                      agg2[(size_t)node * 256 + 128 + c]) + b2[c];
    int g = batch[node];
    atomicAdd(&pooled[(size_t)g * 128 + c], o);
    if (c == 0) atomicAdd(&cnt[g], 1.0f);
}

__global__ void pool_fin_kernel(const float* __restrict__ pooled,
                                const float* __restrict__ cnt,
                                float* __restrict__ out, int ng) {
    int i = blockIdx.x * blockDim.x + threadIdx.x;
    if (i >= ng * 128) return;
    int g = i >> 7;
    float v = pooled[i] / fmaxf(cnt[g], 1.0f);
    out[i] = (v > 0.f) ? v : 0.f;
}

// ---------------------------------------------------------------------------
extern "C" void kernel_launch(void* const* d_in, const int* in_sizes, int n_in,
                              void* d_out, int out_size, void* d_ws,
                              size_t ws_size, hipStream_t stream) {
    const float* x   = (const float*)d_in[0];
    const int*   ei  = (const int*)d_in[1];
    const int*   bat = (const int*)d_in[2];
    const float* W1  = (const float*)d_in[4];
    const float* as1 = (const float*)d_in[5];
    const float* ad1 = (const float*)d_in[6];
    const float* b1  = (const float*)d_in[7];
    const float* W2  = (const float*)d_in[8];
    const float* as2 = (const float*)d_in[9];
    const float* ad2 = (const float*)d_in[10];
    const float* b2  = (const float*)d_in[11];

    const int N  = in_sizes[0] / 128;
    const int E  = in_sizes[1] / 2;
    const int NG = out_size / 128;
    const int ET = E + N;
    (void)n_in; (void)ws_size;

    float* ws = (float*)d_ws;
    size_t o = 0;
    float* xh1   = ws + o; o += (size_t)N * 128;
    float* agg1  = ws + o; o += (size_t)N * 128;   // becomes relu(out1)
    float* xh2   = ws + o; o += (size_t)N * 256;
    float* agg2  = ws + o; o += (size_t)N * 256;
    float* A1s   = ws + o; o += (size_t)N * 2;
    float* A1d   = ws + o; o += (size_t)N * 2;
    float* A2s   = ws + o; o += (size_t)N * 2;
    float* A2d   = ws + o; o += (size_t)N * 2;
    unsigned* mord = (unsigned*)(ws + o); o += (size_t)N * 2;
    float* denom = ws + o; o += (size_t)N * 2;
    float* expe  = ws + o; o += (size_t)ET * 2;
    float* pooled= ws + o; o += (size_t)NG * 128;
    float* cnt   = ws + o; o += (size_t)NG;

    const int gemmBlocks = ((N + 15) / 16 + 7) / 8;     // 8 waves/block
    const unsigned edgeBlocks = (unsigned)((2LL * ET + 255) / 256);

    // ----- layer 1 (heads=2, C=64, concat -> 128) -----
    hipMemsetAsync(agg1,  0, (size_t)N * 128 * sizeof(float), stream);
    hipMemsetAsync(mord,  0, (size_t)N * 2 * sizeof(unsigned), stream);
    hipMemsetAsync(denom, 0, (size_t)N * 2 * sizeof(float), stream);
    gemm_bf16_wmma<<<gemmBlocks, 256, 128 * 128 * 2, stream>>>(x, W1, xh1, N, 128);
    attn_kernel<<<(N + 7) / 8, 256, 0, stream>>>(xh1, as1, ad1, A1s, A1d, N, 64);
    edge_max_kernel<<<edgeBlocks, 256, 0, stream>>>(ei, E, N, A1s, A1d, mord);
    edge_exp_kernel<<<edgeBlocks, 256, 0, stream>>>(ei, E, N, A1s, A1d, mord,
                                                    expe, denom);
    edge_agg_kernel<<<ET, 64, 0, stream>>>(ei, E, N, xh1, expe, denom, agg1, 64);
    {
        long long tot = (long long)N * 128;
        bias_relu_kernel<<<(unsigned)((tot + 255) / 256), 256, 0, stream>>>(
            agg1, b1, tot, 128);
    }

    // ----- layer 2 (heads=2, C=128, mean -> 128) -----
    hipMemsetAsync(agg2,  0, (size_t)N * 256 * sizeof(float), stream);
    hipMemsetAsync(mord,  0, (size_t)N * 2 * sizeof(unsigned), stream);
    hipMemsetAsync(denom, 0, (size_t)N * 2 * sizeof(float), stream);
    gemm_bf16_wmma<<<gemmBlocks, 256, 128 * 256 * 2, stream>>>(agg1, W2, xh2, N, 256);
    attn_kernel<<<(N + 7) / 8, 256, 0, stream>>>(xh2, as2, ad2, A2s, A2d, N, 128);
    edge_max_kernel<<<edgeBlocks, 256, 0, stream>>>(ei, E, N, A2s, A2d, mord);
    edge_exp_kernel<<<edgeBlocks, 256, 0, stream>>>(ei, E, N, A2s, A2d, mord,
                                                    expe, denom);
    edge_agg_kernel<<<ET, 64, 0, stream>>>(ei, E, N, xh2, expe, denom, agg2, 128);

    // ----- head-mean + bias + global mean pool + relu -----
    hipMemsetAsync(pooled, 0, (size_t)NG * 128 * sizeof(float), stream);
    hipMemsetAsync(cnt,    0, (size_t)NG * sizeof(float), stream);
    mean_pool_kernel<<<N, 128, 0, stream>>>(agg2, b2, bat, pooled, cnt, N);
    pool_fin_kernel<<<(NG * 128 + 255) / 256, 256, 0, stream>>>(
        pooled, cnt, (float*)d_out, NG);
}
